// Token_QK_Attention_936302870963
// MI455X (gfx1250) — matrix operand order repaired and checked by reading, not compiled, not run on hardware
//
#include <hip/hip_runtime.h>

typedef __attribute__((ext_vector_type(16))) _Float16 v16h;
typedef __attribute__((ext_vector_type(8)))  _Float16 v8h;
typedef __attribute__((ext_vector_type(4)))  _Float16 v4h;
typedef __attribute__((ext_vector_type(2)))  _Float16 v2h;
typedef __attribute__((ext_vector_type(8)))  float    v8f;

#define TT 4
#define BB 32
#define CC 384
#define NN 256
#define NT 16
#define DT 32
#define CSTEP 128
#define PAD 8
#define NSTAGE (CC / CSTEP)
#define HEADS 8
#define CHD 48

#define XTOT ((size_t)TT * BB * CC * NN)   /* 12,582,912 f16 */
#define WTOT ((size_t)3 * CC * CC)         /* 442,368  f16: rows [q|k|proj] */

__device__ __forceinline__ v16h cat8(v8h lo, v8h hi) {
  return __builtin_shufflevector(lo, hi, 0, 1, 2, 3, 4, 5, 6, 7, 8, 9, 10, 11, 12, 13, 14, 15);
}

__device__ __forceinline__ void async_lds_b128(const _Float16* gsrc, void* ldst) {
  uint32_t lds = (uint32_t)(uintptr_t)ldst;   // low 32 bits of generic ptr = LDS byte offset
  asm volatile("global_load_async_to_lds_b128 %0, %1, off"
               :: "v"(lds), "v"(gsrc) : "memory");
}

__device__ __forceinline__ void wait_async0() {
#if __has_builtin(__builtin_amdgcn_s_wait_asynccnt)
  __builtin_amdgcn_s_wait_asynccnt(0);
#else
  asm volatile("s_wait_asynccnt 0x0" ::: "memory");
#endif
}

// -------- Kernel 0: one-shot f32 -> f16 conversion of x and the stacked weights --------
__global__ __launch_bounds__(256) void cvt_kernel(
    const float* __restrict__ x,
    const float* __restrict__ qw, const float* __restrict__ kw, const float* __restrict__ pw,
    _Float16* __restrict__ xh, _Float16* __restrict__ wst)
{
  size_t i = ((size_t)blockIdx.x * 256 + threadIdx.x) * 4;
  if (i < XTOT) {
    float4 v = *(const float4*)(x + i);
    v4h o = {(_Float16)v.x, (_Float16)v.y, (_Float16)v.z, (_Float16)v.w};
    *(v4h*)(xh + i) = o;
  } else if (i < XTOT + WTOT) {
    size_t j = i - XTOT;
    int row = (int)(j / CC), col = (int)(j % CC);
    const float* src = (row < CC) ? &qw[(size_t)row * CC]
                     : (row < 2 * CC) ? &kw[(size_t)(row - CC) * CC]
                                      : &pw[(size_t)(row - 2 * CC) * CC];
    float4 v = *(const float4*)(src + col);
    v4h o = {(_Float16)v.x, (_Float16)v.y, (_Float16)v.z, (_Float16)v.w};
    *(v4h*)(wst + j) = o;
  }
}

// -------- Kernel 1: fused (q|k) GEMM -> BN -> 4-step LIF -> f16 spikes [T,B,2C,N] --------
__global__ __launch_bounds__(256) void qk_kernel(
    const _Float16* __restrict__ xh, const _Float16* __restrict__ wst,
    const float* __restrict__ qg, const float* __restrict__ qb,
    const float* __restrict__ qm, const float* __restrict__ qv,
    const float* __restrict__ kg, const float* __restrict__ kb,
    const float* __restrict__ km, const float* __restrict__ kv,
    _Float16* __restrict__ spk)
{
  __shared__ __align__(16) _Float16 sW[DT][CSTEP + PAD];
  __shared__ __align__(16) _Float16 sX[TT][NT][CSTEP + PAD];
  __shared__ float sPre[TT][DT][NT];

  const int tid  = threadIdx.x;
  const int lane = tid & 31, wv = tid >> 5;
  const int g = lane >> 4, l = lane & 15;
  const int t = wv & 3, dsub = wv >> 2;

  const int b    = blockIdx.x >> 4;
  const int n0   = (blockIdx.x & 15) * NT;
  const int dblk = blockIdx.y;                  // 0..23 over stacked [q|k] rows
  const bool isk = dblk >= 12;
  const int d0   = (isk ? dblk - 12 : dblk) * DT;
  const int dstack0 = dblk * DT;
  const _Float16* __restrict__ wrow = wst + (size_t)dstack0 * CC;
  const float* __restrict__ gm = isk ? kg : qg;
  const float* __restrict__ bt = isk ? kb : qb;
  const float* __restrict__ mn = isk ? km : qm;
  const float* __restrict__ vr = isk ? kv : qv;

  v8f acc = {};
  for (int s = 0; s < NSTAGE; ++s) {
    const int c0 = s * CSTEP;
    __syncthreads();
    // W tile: async global->LDS DMA, 16B per lane, layout already A-fragment friendly.
#pragma unroll
    for (int i = 0; i < 2; ++i) {
      int q = tid + i * 256;                    // 512 chunks of 8 halves
      int row = q >> 4, col8 = (q & 15) * 8;
      async_lds_b128(wrow + (size_t)row * CC + c0 + col8, &sW[row][col8]);
    }
    // X tile: registers + transposed packed stores (B fragment needs [n][c]).
#pragma unroll
    for (int i = 0; i < 2; ++i) {
      int u = tid + i * 256;                    // 512 units: (t,ccpair,nchunk)
      int nc = u & 1, cp = (u >> 1) & 63, t2 = u >> 7;
      int cc = cp * 2;
      const _Float16* gp = xh + (((size_t)t2 * BB + b) * CC + c0 + cc) * NN + n0 + nc * 8;
      v8h r0 = *(const v8h*)gp;
      v8h r1 = *(const v8h*)(gp + NN);
#pragma unroll
      for (int j = 0; j < 8; ++j) {
        v2h p = {r0[j], r1[j]};
        *(v2h*)&sX[t2][nc * 8 + j][cc] = p;
      }
    }
    wait_async0();
    __syncthreads();
#pragma unroll
    for (int kk = 0; kk < CSTEP; kk += 32) {
      v8h alo = *(const v8h*)&sW[dsub * 16 + l][kk + g * 8];
      v8h ahi = *(const v8h*)&sW[dsub * 16 + l][kk + 16 + g * 8];
      v8h blo = *(const v8h*)&sX[t][l][kk + g * 8];
      v8h bhi = *(const v8h*)&sX[t][l][kk + 16 + g * 8];
      acc = __builtin_amdgcn_wmma_f32_16x16x32_f16(false, cat8(alo, ahi),
                                                   false, cat8(blo, bhi),
                                                   (short)0, acc, false, false);
    }
  }

  // BN on accumulator; stage all 4 timesteps to LDS for the LIF scan.
#pragma unroll
  for (int r = 0; r < 8; ++r) {
    int dl = dsub * 16 + g * 8 + r;
    int dg = d0 + dl;
    float sc = gm[dg] * rsqrtf(vr[dg] + 1e-5f);
    float sh = bt[dg] - mn[dg] * sc;
    sPre[t][dl][l] = acc[r] * sc + sh;
  }
  __syncthreads();

  // LIF: h = (v+x)/2, spike = h>=1, hard reset to 0.
#pragma unroll
  for (int j = 0; j < 2; ++j) {
    int idx = tid + j * 256;                    // [DT][NT]
    int dl = idx >> 4, n = idx & 15;
    float v = 0.f;
#pragma unroll
    for (int ts = 0; ts < TT; ++ts) {
      float h = 0.5f * (v + sPre[ts][dl][n]);
      float sp = (h >= 1.0f) ? 1.0f : 0.0f;
      v = h * (1.0f - sp);
      spk[(((size_t)ts * BB + b) * (2 * CC) + dstack0 + dl) * NN + n0 + n] = (_Float16)sp;
    }
  }
}

// -------- Kernel 2: head-sum of q spikes -> attn LIF (thr 0.5) -> y = attn * k --------
__global__ __launch_bounds__(256) void attn_kernel(
    const _Float16* __restrict__ spk, _Float16* __restrict__ y)
{
  const int b = blockIdx.x >> 3;
  const int h = blockIdx.x & 7;
  const int n = threadIdx.x;

  float v = 0.f;
  float s_t[TT];
#pragma unroll
  for (int t = 0; t < TT; ++t) {
    size_t qbase = (((size_t)t * BB + b) * (2 * CC) + h * CHD) * NN + n;
    float sum = 0.f;
#pragma unroll 8
    for (int j = 0; j < CHD; ++j) sum += (float)spk[qbase + (size_t)j * NN];
    float hh = 0.5f * (v + sum);
    float sp = (hh >= 0.5f) ? 1.0f : 0.0f;
    v = hh * (1.0f - sp);
    s_t[t] = sp;
  }
#pragma unroll
  for (int t = 0; t < TT; ++t) {
    size_t kbase = (((size_t)t * BB + b) * (2 * CC) + CC + h * CHD) * NN + n;
    size_t ybase = (((size_t)t * BB + b) * CC + h * CHD) * NN + n;
#pragma unroll 8
    for (int j = 0; j < CHD; ++j)
      y[ybase + (size_t)j * NN] = (_Float16)(s_t[t] * (float)spk[kbase + (size_t)j * NN]);
  }
}

// -------- Kernel 3: proj GEMM (+bias) -> BN -> LIF -> f32 out [T,B,C,N] --------
__global__ __launch_bounds__(256) void proj_kernel(
    const _Float16* __restrict__ y, const _Float16* __restrict__ wst,
    const float* __restrict__ pg, const float* __restrict__ pb,
    const float* __restrict__ pm, const float* __restrict__ pv,
    const float* __restrict__ pbias,
    float* __restrict__ out)
{
  __shared__ __align__(16) _Float16 sW[DT][CSTEP + PAD];
  __shared__ __align__(16) _Float16 sX[TT][NT][CSTEP + PAD];
  __shared__ float sPre[TT][DT][NT];

  const int tid  = threadIdx.x;
  const int lane = tid & 31, wv = tid >> 5;
  const int g = lane >> 4, l = lane & 15;
  const int t = wv & 3, dsub = wv >> 2;

  const int b  = blockIdx.x >> 4;
  const int n0 = (blockIdx.x & 15) * NT;
  const int d0 = blockIdx.y * DT;
  const _Float16* __restrict__ wrow = wst + (size_t)(2 * CC + d0) * CC;  // proj rows

  v8f acc = {};
  for (int s = 0; s < NSTAGE; ++s) {
    const int c0 = s * CSTEP;
    __syncthreads();
#pragma unroll
    for (int i = 0; i < 2; ++i) {
      int q = tid + i * 256;
      int row = q >> 4, col8 = (q & 15) * 8;
      async_lds_b128(wrow + (size_t)row * CC + c0 + col8, &sW[row][col8]);
    }
#pragma unroll
    for (int i = 0; i < 2; ++i) {
      int u = tid + i * 256;
      int nc = u & 1, cp = (u >> 1) & 63, t2 = u >> 7;
      int cc = cp * 2;
      const _Float16* gp = y + (((size_t)t2 * BB + b) * CC + c0 + cc) * NN + n0 + nc * 8;
      v8h r0 = *(const v8h*)gp;
      v8h r1 = *(const v8h*)(gp + NN);
#pragma unroll
      for (int j = 0; j < 8; ++j) {
        v2h p = {r0[j], r1[j]};
        *(v2h*)&sX[t2][nc * 8 + j][cc] = p;
      }
    }
    wait_async0();
    __syncthreads();
#pragma unroll
    for (int kk = 0; kk < CSTEP; kk += 32) {
      v8h alo = *(const v8h*)&sW[dsub * 16 + l][kk + g * 8];
      v8h ahi = *(const v8h*)&sW[dsub * 16 + l][kk + 16 + g * 8];
      v8h blo = *(const v8h*)&sX[t][l][kk + g * 8];
      v8h bhi = *(const v8h*)&sX[t][l][kk + 16 + g * 8];
      acc = __builtin_amdgcn_wmma_f32_16x16x32_f16(false, cat8(alo, ahi),
                                                   false, cat8(blo, bhi),
                                                   (short)0, acc, false, false);
    }
  }

  // bias + BN folded: val = acc*sc + (bias*sc + shift)
#pragma unroll
  for (int r = 0; r < 8; ++r) {
    int dl = dsub * 16 + g * 8 + r;
    int dg = d0 + dl;
    float sc = pg[dg] * rsqrtf(pv[dg] + 1e-5f);
    float sh = pb[dg] - pm[dg] * sc + pbias[dg] * sc;
    sPre[t][dl][l] = acc[r] * sc + sh;
  }
  __syncthreads();

#pragma unroll
  for (int j = 0; j < 2; ++j) {
    int idx = tid + j * 256;
    int dl = idx >> 4, n = idx & 15;
    float v = 0.f;
#pragma unroll
    for (int ts = 0; ts < TT; ++ts) {
      float h = 0.5f * (v + sPre[ts][dl][n]);
      float sp = (h >= 1.0f) ? 1.0f : 0.0f;
      v = h * (1.0f - sp);
      out[(((size_t)ts * BB + b) * CC + d0 + dl) * NN + n0 + n] = sp;
    }
  }
}

extern "C" void kernel_launch(void* const* d_in, const int* in_sizes, int n_in,
                              void* d_out, int out_size, void* d_ws, size_t ws_size,
                              hipStream_t stream) {
  (void)in_sizes; (void)n_in; (void)out_size; (void)ws_size;
  const float* x     = (const float*)d_in[0];
  const float* qw    = (const float*)d_in[1];
  const float* qg    = (const float*)d_in[2];
  const float* qb    = (const float*)d_in[3];
  const float* qm    = (const float*)d_in[4];
  const float* qv    = (const float*)d_in[5];
  const float* kw    = (const float*)d_in[6];
  const float* kg    = (const float*)d_in[7];
  const float* kb    = (const float*)d_in[8];
  const float* km    = (const float*)d_in[9];
  const float* kv    = (const float*)d_in[10];
  const float* pw    = (const float*)d_in[11];
  const float* pg    = (const float*)d_in[12];
  const float* pb    = (const float*)d_in[13];
  const float* pm    = (const float*)d_in[14];
  const float* pv    = (const float*)d_in[15];
  const float* pbias = (const float*)d_in[16];

  _Float16* xh  = (_Float16*)d_ws;                     // [T,B,C,N]  f16 copy of x
  _Float16* wst = xh + XTOT;                           // [3C, C]    f16 stacked weights
  _Float16* spk = wst + WTOT;                          // [T,B,2C,N] f16 spikes (q|k)
  _Float16* ybf = spk + (size_t)TT * BB * 2 * CC * NN; // [T,B,C,N]  f16 y
  float* out = (float*)d_out;

  int cvt_blocks = (int)((XTOT + WTOT) / (256 * 4));   // exact: 12720
  cvt_kernel<<<dim3(cvt_blocks), 256, 0, stream>>>(x, qw, kw, pw, xh, wst);
  qk_kernel<<<dim3(BB * (NN / NT), 24), 256, 0, stream>>>(
      xh, wst, qg, qb, qm, qv, kg, kb, km, kv, spk);
  attn_kernel<<<dim3(BB * HEADS), 256, 0, stream>>>(spk, ybf);
  proj_kernel<<<dim3(BB * (NN / NT), 12), 256, 0, stream>>>(
      ybf, wst, pg, pb, pm, pv, pbias, out);
}
